// LSTMModel2_90623809945617
// MI455X (gfx1250) — matrix operand order repaired and checked by reading, not compiled, not run on hardware
//
#include <hip/hip_runtime.h>
#include <hip/hip_bf16.h>

typedef __attribute__((ext_vector_type(16))) __bf16       v16bf;
typedef __attribute__((ext_vector_type(8)))  float        v8f;
typedef __attribute__((ext_vector_type(4)))  unsigned int u32x4;

#define B_DIM 256
#define T_DIM 256
#define D_DIM 128
#define H_DIM 1024
#define O_DIM 128

union Frag16 { v16bf v; u32x4 u[2]; };

// Force global (addrspace 1) vector loads: never let these become flat_load.
typedef const __attribute__((address_space(1))) u32x4* gptr_u4;

__device__ __forceinline__ u32x4 gload16(const __bf16* base, unsigned elemOff) {
    return *(gptr_u4)(base + elemOff);
}

__device__ __forceinline__ void load_afrag(Frag16& a, const __bf16* A, unsigned off) {
    a.u[0] = gload16(A, off);
    a.u[1] = gload16(A, off + 8);
}

__device__ __forceinline__ void load_bfrag(Frag16& b, const __bf16* W, unsigned off) {
    b.u[0] = gload16(W, off);
    b.u[1] = gload16(W, off + 8);
}

__device__ __forceinline__ float sigmoidf_(float x) {
    return __builtin_amdgcn_rcpf(1.0f + __expf(-x));
}
__device__ __forceinline__ float tanhf_(float x) {
    // tanh(x) = 1 - 2/(e^{2x}+1)  (v_exp + v_rcp, no ocml branches)
    return 1.0f - 2.0f * __builtin_amdgcn_rcpf(1.0f + __expf(2.0f * x));
}

// One LSTM cell update over a 64(row) x 64(h-col) block tile.
// gates = A1 @ W1^T + A2 @ W2^T + bias; pointwise -> c_io (f32, in place), h_out (bf16).
// Block: 256 threads = 8 waves. Wave w: rh = w&1 (row half, 32 rows), nt = w>>1
// (16-col tile). Each wave: 2 row-tiles x 4 gates = 8 WMMA accumulators.
// Every B fragment feeds 2 WMMAs; every A fragment feeds 4.
// All strides/trip-counts are compile-time constants; offsets are 32-bit.
template <int K1, int LDA1>
__device__ __forceinline__ void lstm_cell_body(
    const __bf16* __restrict__ A1, const __bf16* __restrict__ W1,
    const __bf16* __restrict__ A2, const __bf16* __restrict__ W2,
    const float* __restrict__ bias,
    float* __restrict__ c_io,
    __bf16* __restrict__ h_out,
    int bx, int by)
{
    const int tid  = threadIdx.x;
    const int wave = tid >> 5;
    const int lane = tid & 31;
    const int lr   = lane & 15;      // row-in-tile (A) / col-in-tile (B,C,D)
    const int hi   = lane >> 4;      // lane half
    const int rh   = wave & 1;       // row half of the 64-row block tile
    const int nt   = wave >> 1;      // 16-col tile within 64 cols
    const int row0 = bx * 64 + rh * 32;
    const int col0 = by * 64 + nt * 16;

    v8f acc[4][2];
#pragma unroll
    for (int g = 0; g < 4; ++g) {
        const float bv = bias[g * H_DIM + col0 + lr];
#pragma unroll
        for (int r = 0; r < 8; ++r) { acc[g][0][r] = bv; acc[g][1][r] = bv; }
    }

    // ---- K1 loop: input projection (K1 = 128 or 1024, fully constant) ----
    {
        const unsigned aBase = (unsigned)(row0 + lr) * LDA1 + hi * 16;
        const unsigned wBase = (unsigned)(col0 + lr) * K1 + hi * 16;
#pragma unroll 4
        for (int k0 = 0; k0 < K1; k0 += 32) {
            Frag16 a0, a1, b0, b1, b2, b3;
            load_afrag(a0, A1, aBase + k0);
            load_afrag(a1, A1, aBase + 16u * LDA1 + k0);
            load_bfrag(b0, W1, wBase + k0);
            load_bfrag(b1, W1, wBase + 1u * H_DIM * K1 + k0);
            load_bfrag(b2, W1, wBase + 2u * H_DIM * K1 + k0);
            load_bfrag(b3, W1, wBase + 3u * H_DIM * K1 + k0);
            acc[0][0] = __builtin_amdgcn_wmma_f32_16x16x32_bf16(false, a0.v, false, b0.v, (short)0, acc[0][0], false, false);
            acc[0][1] = __builtin_amdgcn_wmma_f32_16x16x32_bf16(false, a1.v, false, b0.v, (short)0, acc[0][1], false, false);
            acc[1][0] = __builtin_amdgcn_wmma_f32_16x16x32_bf16(false, a0.v, false, b1.v, (short)0, acc[1][0], false, false);
            acc[1][1] = __builtin_amdgcn_wmma_f32_16x16x32_bf16(false, a1.v, false, b1.v, (short)0, acc[1][1], false, false);
            acc[2][0] = __builtin_amdgcn_wmma_f32_16x16x32_bf16(false, a0.v, false, b2.v, (short)0, acc[2][0], false, false);
            acc[2][1] = __builtin_amdgcn_wmma_f32_16x16x32_bf16(false, a1.v, false, b2.v, (short)0, acc[2][1], false, false);
            acc[3][0] = __builtin_amdgcn_wmma_f32_16x16x32_bf16(false, a0.v, false, b3.v, (short)0, acc[3][0], false, false);
            acc[3][1] = __builtin_amdgcn_wmma_f32_16x16x32_bf16(false, a1.v, false, b3.v, (short)0, acc[3][1], false, false);
        }
    }

    // ---- K2 loop: recurrent projection (K2 = H = 1024) ----
    {
        const unsigned aBase = (unsigned)(row0 + lr) * H_DIM + hi * 16;
        const unsigned wBase = (unsigned)(col0 + lr) * H_DIM + hi * 16;
#pragma unroll 2
        for (int k0 = 0; k0 < H_DIM; k0 += 32) {
            Frag16 a0, a1, b0, b1, b2, b3;
            load_afrag(a0, A2, aBase + k0);
            load_afrag(a1, A2, aBase + 16u * H_DIM + k0);
            load_bfrag(b0, W2, wBase + k0);
            load_bfrag(b1, W2, wBase + 1u * H_DIM * H_DIM + k0);
            load_bfrag(b2, W2, wBase + 2u * H_DIM * H_DIM + k0);
            load_bfrag(b3, W2, wBase + 3u * H_DIM * H_DIM + k0);
            acc[0][0] = __builtin_amdgcn_wmma_f32_16x16x32_bf16(false, a0.v, false, b0.v, (short)0, acc[0][0], false, false);
            acc[0][1] = __builtin_amdgcn_wmma_f32_16x16x32_bf16(false, a1.v, false, b0.v, (short)0, acc[0][1], false, false);
            acc[1][0] = __builtin_amdgcn_wmma_f32_16x16x32_bf16(false, a0.v, false, b1.v, (short)0, acc[1][0], false, false);
            acc[1][1] = __builtin_amdgcn_wmma_f32_16x16x32_bf16(false, a1.v, false, b1.v, (short)0, acc[1][1], false, false);
            acc[2][0] = __builtin_amdgcn_wmma_f32_16x16x32_bf16(false, a0.v, false, b2.v, (short)0, acc[2][0], false, false);
            acc[2][1] = __builtin_amdgcn_wmma_f32_16x16x32_bf16(false, a1.v, false, b2.v, (short)0, acc[2][1], false, false);
            acc[3][0] = __builtin_amdgcn_wmma_f32_16x16x32_bf16(false, a0.v, false, b3.v, (short)0, acc[3][0], false, false);
            acc[3][1] = __builtin_amdgcn_wmma_f32_16x16x32_bf16(false, a1.v, false, b3.v, (short)0, acc[3][1], false, false);
        }
    }

    // ---- Pointwise LSTM update (gate order: i, f, g, o) ----
    const int col = col0 + lr;
#pragma unroll
    for (int mm = 0; mm < 2; ++mm) {
#pragma unroll
        for (int r = 0; r < 8; ++r) {
            const int row = row0 + mm * 16 + r + 8 * hi;
            const unsigned idx = (unsigned)row * H_DIM + col;
            const float iv = acc[0][mm][r];
            const float fv = acc[1][mm][r];
            const float gv = acc[2][mm][r];
            const float ov = acc[3][mm][r];
            const float cp = c_io[idx];
            const float cn = sigmoidf_(fv) * cp + sigmoidf_(iv) * tanhf_(gv);
            const float hn = sigmoidf_(ov) * tanhf_(cn);
            c_io[idx]  = cn;
            h_out[idx] = (__bf16)hn;
        }
    }
}

// Pipelined pair: blockIdx.z==0 -> layer0 at step k (K1=D, A1 strided T*D),
// blockIdx.z==1 -> layer1 at step k-1 (K1=H). Both depend only on buffers from
// the previous launch -> both layers' GEMMs run concurrently in one launch.
__global__ __launch_bounds__(256) void lstm_pair_wmma(
    int do0, int do1,
    const __bf16* __restrict__ A1_0, const __bf16* __restrict__ W1_0,
    const __bf16* __restrict__ A2_0, const __bf16* __restrict__ W2_0,
    const float* __restrict__ bias_0, float* __restrict__ c_0, __bf16* __restrict__ h_0,
    const __bf16* __restrict__ A1_1, const __bf16* __restrict__ W1_1,
    const __bf16* __restrict__ A2_1, const __bf16* __restrict__ W2_1,
    const float* __restrict__ bias_1, float* __restrict__ c_1, __bf16* __restrict__ h_1)
{
    if (blockIdx.z == 0) {
        if (!do0) return;
        lstm_cell_body<D_DIM, T_DIM * D_DIM>(A1_0, W1_0, A2_0, W2_0,
                                             bias_0, c_0, h_0, blockIdx.x, blockIdx.y);
    } else {
        if (!do1) return;
        lstm_cell_body<H_DIM, H_DIM>(A1_1, W1_1, A2_1, W2_1,
                                     bias_1, c_1, h_1, blockIdx.x, blockIdx.y);
    }
}

// out = A @ W^T + bias : [256 x 128] = [256 x 1024] x [128 x 1024]^T
__global__ __launch_bounds__(256) void linear_wmma(
    const __bf16* __restrict__ A,      // [B x H] bf16
    const __bf16* __restrict__ W,      // [O x H] bf16
    const float* __restrict__ bias,    // [O] f32
    float* __restrict__ out)           // [B x O] f32
{
    const int tid  = threadIdx.x;
    const int wave = tid >> 5;
    const int lane = tid & 31;
    const int lr   = lane & 15;
    const int hi   = lane >> 4;
    const int m    = wave & 1;
    const int nt   = wave >> 1;
    const int row0 = blockIdx.x * 32 + m * 16;
    const int col0 = blockIdx.y * 64 + nt * 16;

    v8f acc;
    {
        const float bv = bias[col0 + lr];
#pragma unroll
        for (int r = 0; r < 8; ++r) acc[r] = bv;
    }

    const unsigned aBase = (unsigned)(row0 + lr) * H_DIM + hi * 16;
    const unsigned wBase = (unsigned)(col0 + lr) * H_DIM + hi * 16;
#pragma unroll 2
    for (int k0 = 0; k0 < H_DIM; k0 += 32) {
        Frag16 a, b;
        load_afrag(a, A, aBase + k0);
        load_bfrag(b, W, wBase + k0);
        acc = __builtin_amdgcn_wmma_f32_16x16x32_bf16(
            false, a.v, false, b.v, (short)0, acc, false, false);
    }

    const int col = col0 + lr;
#pragma unroll
    for (int r = 0; r < 8; ++r) {
        const int row = row0 + r + 8 * hi;
        out[(unsigned)row * O_DIM + col] = acc[r];
    }
}

__global__ void f32_to_bf16_kernel(const float* __restrict__ s, __bf16* __restrict__ d, int n) {
    int i = blockIdx.x * 256 + threadIdx.x;
    if (i < n) d[i] = (__bf16)s[i];
}

__global__ void copy_f32_kernel(const float* __restrict__ s, float* __restrict__ d, int n) {
    int i = blockIdx.x * 256 + threadIdx.x;
    if (i < n) d[i] = s[i];
}

__global__ void add_f32_kernel(const float* __restrict__ a, const float* __restrict__ b,
                               float* __restrict__ d, int n) {
    int i = blockIdx.x * 256 + threadIdx.x;
    if (i < n) d[i] = a[i] + b[i];
}

extern "C" void kernel_launch(void* const* d_in, const int* in_sizes, int n_in,
                              void* d_out, int out_size, void* d_ws, size_t ws_size,
                              hipStream_t stream) {
    const float* x     = (const float*)d_in[0];   // [B,T,D]
    const float* h0    = (const float*)d_in[1];   // [B,H]
    const float* c0    = (const float*)d_in[2];   // [B,H]
    const float* W_ih0 = (const float*)d_in[3];   // [4H,D]
    const float* W_hh0 = (const float*)d_in[4];   // [4H,H]
    const float* b_ih0 = (const float*)d_in[5];
    const float* b_hh0 = (const float*)d_in[6];
    const float* W_ih1 = (const float*)d_in[7];   // [4H,H]
    const float* W_hh1 = (const float*)d_in[8];   // [4H,H]
    const float* b_ih1 = (const float*)d_in[9];
    const float* b_hh1 = (const float*)d_in[10];
    const float* W_out = (const float*)d_in[11];  // [O,H]
    const float* b_out = (const float*)d_in[12];  // [O]

    // ---- workspace carve-up (256B aligned) ----
    char* w = (char*)d_ws;
    auto alloc = [&](size_t bytes) {
        char* p = w;
        w += (bytes + 255) & ~(size_t)255;
        return p;
    };
    const int nX    = B_DIM * T_DIM * D_DIM;   // 8388608
    const int nWih0 = 4 * H_DIM * D_DIM;       // 524288
    const int nWhh  = 4 * H_DIM * H_DIM;       // 4194304
    const int nWout = O_DIM * H_DIM;           // 131072
    const int nBH   = B_DIM * H_DIM;           // 262144

    __bf16* xb    = (__bf16*)alloc((size_t)nX * 2);
    __bf16* Wih0b = (__bf16*)alloc((size_t)nWih0 * 2);
    __bf16* Whh0b = (__bf16*)alloc((size_t)nWhh * 2);
    __bf16* Wih1b = (__bf16*)alloc((size_t)nWhh * 2);
    __bf16* Whh1b = (__bf16*)alloc((size_t)nWhh * 2);
    __bf16* Woutb = (__bf16*)alloc((size_t)nWout * 2);
    __bf16* ha[2] = { (__bf16*)alloc((size_t)nBH * 2), (__bf16*)alloc((size_t)nBH * 2) };
    __bf16* hb[2] = { (__bf16*)alloc((size_t)nBH * 2), (__bf16*)alloc((size_t)nBH * 2) };
    float*  ca    = (float*)alloc((size_t)nBH * 4);
    float*  cb    = (float*)alloc((size_t)nBH * 4);
    float*  bias0 = (float*)alloc(4 * H_DIM * 4);
    float*  bias1 = (float*)alloc(4 * H_DIM * 4);

    auto blocks = [](int n) { return (n + 255) / 256; };

    // ---- init / precision conversion (redone every call: deterministic) ----
    f32_to_bf16_kernel<<<blocks(nX),    256, 0, stream>>>(x,     xb,    nX);
    f32_to_bf16_kernel<<<blocks(nWih0), 256, 0, stream>>>(W_ih0, Wih0b, nWih0);
    f32_to_bf16_kernel<<<blocks(nWhh),  256, 0, stream>>>(W_hh0, Whh0b, nWhh);
    f32_to_bf16_kernel<<<blocks(nWhh),  256, 0, stream>>>(W_ih1, Wih1b, nWhh);
    f32_to_bf16_kernel<<<blocks(nWhh),  256, 0, stream>>>(W_hh1, Whh1b, nWhh);
    f32_to_bf16_kernel<<<blocks(nWout), 256, 0, stream>>>(W_out, Woutb, nWout);
    f32_to_bf16_kernel<<<blocks(nBH),   256, 0, stream>>>(h0, ha[0], nBH);
    f32_to_bf16_kernel<<<blocks(nBH),   256, 0, stream>>>(h0, hb[0], nBH);
    copy_f32_kernel   <<<blocks(nBH),   256, 0, stream>>>(c0, ca, nBH);
    copy_f32_kernel   <<<blocks(nBH),   256, 0, stream>>>(c0, cb, nBH);
    add_f32_kernel    <<<blocks(4*H_DIM), 256, 0, stream>>>(b_ih0, b_hh0, bias0, 4*H_DIM);
    add_f32_kernel    <<<blocks(4*H_DIM), 256, 0, stream>>>(b_ih1, b_hh1, bias1, 4*H_DIM);

    // ---- skewed sequential scan: launch k does layer0(k) and layer1(k-1) ----
    // layer0(k):  reads h_a(k-1)=ha[k&1], writes h_a(k)=ha[(k+1)&1], c_a in place
    // layer1(k-1): reads h_a(k-1)=ha[k&1], h_b(k-2)=hb[(k+1)&1],
    //              writes h_b(k-1)=hb[k&1], c_b in place
    const dim3 pairGrid(B_DIM / 64, H_DIM / 64, 2);   // (4, 16, 2)
    for (int k = 0; k <= T_DIM; ++k) {
        const int do0 = (k < T_DIM) ? 1 : 0;
        const int do1 = (k >= 1) ? 1 : 0;
        const int tx = (k < T_DIM) ? k : (T_DIM - 1);
        const __bf16* ha_in  = ha[k & 1];
        __bf16*       ha_out = ha[(k + 1) & 1];
        const __bf16* hb_in  = hb[(k + 1) & 1];
        __bf16*       hb_out = hb[k & 1];

        lstm_pair_wmma<<<pairGrid, 256, 0, stream>>>(
            do0, do1,
            // layer 0, step k: gates = x_k @ W_ih0^T + h_a @ W_hh0^T + bias0
            xb + (size_t)tx * D_DIM, Wih0b, ha_in, Whh0b, bias0, ca, ha_out,
            // layer 1, step k-1: gates = h_a(k-1) @ W_ih1^T + h_b(k-2) @ W_hh1^T + bias1
            ha_in, Wih1b, hb_in, Whh1b, bias1, cb, hb_out);
    }

    // ---- output projection: h_b(T-1) @ W_out^T + b_out ----
    // Last layer1 write happened at launch k=T into hb[T&1] == hb[0].
    linear_wmma<<<dim3(B_DIM / 32, O_DIM / 64), 256, 0, stream>>>(
        hb[0], Woutb, b_out, (float*)d_out);
}
